// ArcNegFace_52072183497440
// MI455X (gfx1250) — compile-verified, matrix operand order checked by reading, not run on hardware
//
#include <hip/hip_runtime.h>

// ---------------------------------------------------------------------------
// ArcNegFace for MI455X (gfx1250, wave32).
// cos = normalize(feats) @ normalize(weight)^T  [512 x 100000], K=512
// Roofline: ~410MB HBM traffic (weight + out) -> ~18us @ 23.3TB/s.
// AI ~128 FLOP/B => need PFLOP-class matmul => bf16 WMMA (f32 accumulate).
// A-tile staging uses GLOBAL_LOAD_ASYNC_TO_LDS_B128 (ASYNCcnt) — CDNA5 path.
// ---------------------------------------------------------------------------

typedef __bf16 bf16_t;
typedef __bf16 v16bf __attribute__((ext_vector_type(16)));
typedef __bf16 v8bf  __attribute__((ext_vector_type(8)));
typedef float  v8f   __attribute__((ext_vector_type(8)));

#define B_DIM 512
#define D_DIM 512
#define C_DIM 100000

#define NTILE 128
#define MTILE 128
#define KPAD  (D_DIM + 8)   // 520 bf16 per LDS row: 1040B = 260 dwords (==4 mod 64 banks)

// margin constants (MARGIN = 0.5)
#define COS_M   0.8775825618903728f   // cos(0.5)
#define SIN_M   0.4794255386042030f   // sin(0.5)
#define THRESH (-0.8775825618903728f) // cos(pi - 0.5)
#define MM_C    0.2397127693021015f   // sin(pi-0.5)*0.5
#define SCALE_F 64.0f
#define ALPHA_F 1.2f
#define INV_SIGMA 0.5f                // 1/SIGMA, SIGMA=2

// ---------------- kernel 1: inverse L2 norm of each weight row -------------
__global__ __launch_bounds__(256)
void wnorm_k(const float* __restrict__ W, float* __restrict__ invw) {
    const int wave = threadIdx.x >> 5;
    const int lane = threadIdx.x & 31;
    const int row  = blockIdx.x * 8 + wave;
    if (row >= C_DIM) return;
    const float4* Wr = (const float4*)(W + (size_t)row * D_DIM);
    float s = 0.f;
#pragma unroll
    for (int i = 0; i < 4; ++i) {           // 32 lanes * 4 iters * float4 = 512
        float4 v = Wr[lane + i * 32];
        s += v.x * v.x + v.y * v.y + v.z * v.z + v.w * v.w;
    }
#pragma unroll
    for (int off = 16; off > 0; off >>= 1) s += __shfl_xor(s, off, 32);
    if (lane == 0) invw[row] = rsqrtf(s);
}

// -------- kernel 2: normalize feats -> bf16 ex; per-row margin a_t ---------
__global__ __launch_bounds__(256)
void fprep_k(const float* __restrict__ feats, const float* __restrict__ W,
             const int* __restrict__ labels, const float* __restrict__ invw,
             bf16_t* __restrict__ exb, float* __restrict__ a_t) {
    const int b = blockIdx.x;
    const int t = threadIdx.x;
    __shared__ float red[256];

    const float* fr = feats + (size_t)b * D_DIM;
    float f0 = fr[t], f1 = fr[t + 256];
    red[t] = f0 * f0 + f1 * f1;
    __syncthreads();
    for (int off = 128; off > 0; off >>= 1) {
        if (t < off) red[t] += red[t + off];
        __syncthreads();
    }
    const float inv = rsqrtf(red[0]);

    exb[(size_t)b * D_DIM + t]       = (bf16_t)(f0 * inv);
    exb[(size_t)b * D_DIM + t + 256] = (bf16_t)(f1 * inv);

    const int lab = labels[b];
    const float* wr = W + (size_t)lab * D_DIM;
    float d = f0 * inv * wr[t] + f1 * inv * wr[t + 256];
    __syncthreads();
    red[t] = d;
    __syncthreads();
    for (int off = 128; off > 0; off >>= 1) {
        if (t < off) red[t] += red[t + off];
        __syncthreads();
    }
    if (t == 0) {
        float tgt = red[0] * invw[lab];
        float at;
        if (tgt > THRESH) {
            float tc = fminf(fmaxf(tgt, -1.f), 1.f);
            at = tc * COS_M - sqrtf(fmaxf(0.f, 1.f - tc * tc)) * SIN_M;
        } else {
            at = tgt - MM_C;
        }
        a_t[b] = at;
    }
}

// ---- branchless fused ArcNegFace epilogue; GUARD only for last N tile -----
template <bool GUARD>
__device__ __forceinline__ void arc_epilogue(
    const v8f (&acc)[4][2], const float* __restrict__ sAt,
    const int* __restrict__ sLb, int n0, int waveM, int waveN, int lhi,
    int llo, int m0g, float* __restrict__ out) {
#pragma unroll
    for (int tm = 0; tm < 4; ++tm) {
        const int mlb = waveM * 64 + tm * 16 + lhi * 8;   // 8-aligned
        const float4 at0 = *(const float4*)(sAt + mlb);
        const float4 at1 = *(const float4*)(sAt + mlb + 4);
        const int4   lb0 = *(const int4*)(sLb + mlb);
        const int4   lb1 = *(const int4*)(sLb + mlb + 4);
        const float atv[8] = {at0.x, at0.y, at0.z, at0.w,
                              at1.x, at1.y, at1.z, at1.w};
        const int   lbv[8] = {lb0.x, lb0.y, lb0.z, lb0.w,
                              lb1.x, lb1.y, lb1.z, lb1.w};
#pragma unroll
        for (int tn = 0; tn < 2; ++tn) {
            const int ng = n0 + waveN * 32 + tn * 16 + llo;
#pragma unroll
            for (int r = 0; r < 8; ++r) {
                const float at   = atv[r];
                const float cosv = acc[tm][tn][r];
                const float dd   = cosv - at;
                const float rw   = ALPHA_F * __expf(-(dd * dd) * INV_SIGMA);
                const float neg  = SCALE_F * (rw * cosv + rw - 1.f);
                const float val  = (ng == lbv[r]) ? SCALE_F * at : neg; // cndmask
                if (!GUARD || ng < C_DIM)
                    out[(size_t)(m0g + mlb + r) * C_DIM + ng] = val;
            }
        }
    }
}

// -------- kernel 3: bf16 WMMA GEMM + fused ArcNegFace epilogue -------------
// One WG (8 waves) per 128-wide column tile; weight tile read from HBM ONCE,
// normalized+converted to bf16 in LDS, reused across all 4 row blocks.
__global__ __launch_bounds__(256)
void arcneg_gemm_k(const float* __restrict__ W, const float* __restrict__ invw,
                   const bf16_t* __restrict__ exb, const float* __restrict__ a_t,
                   const int* __restrict__ labels, float* __restrict__ out) {
    extern __shared__ char smem[];
    bf16_t* Bs  = (bf16_t*)smem;                              // NTILE * KPAD bf16
    bf16_t* As  = (bf16_t*)(smem + (size_t)NTILE * KPAD * 2); // MTILE * KPAD bf16
    float*  sAt = (float*)(smem + (size_t)(NTILE + MTILE) * KPAD * 2);
    int*    sLb = (int*)(sAt + MTILE);

    const int tid = threadIdx.x;
    const int n0  = blockIdx.x * NTILE;
    const bool full_tile = (n0 + NTILE) <= C_DIM;

    // ---- stage B: load f32 weight tile, normalize, convert to bf16 in LDS ----
    for (int i = tid; i < NTILE * (D_DIM / 4); i += 256) {
        const int n  = i >> 7;              // D_DIM/4 == 128
        const int k4 = (i & 127) << 2;
        const int ng = n0 + n;
        float4 v = make_float4(0.f, 0.f, 0.f, 0.f);
        float  s = 0.f;
        if (ng < C_DIM) {
            v = *(const float4*)(W + (size_t)ng * D_DIM + k4);
            s = invw[ng];
        }
        bf16_t* dst = Bs + n * KPAD + k4;
        dst[0] = (bf16_t)(v.x * s);
        dst[1] = (bf16_t)(v.y * s);
        dst[2] = (bf16_t)(v.z * s);
        dst[3] = (bf16_t)(v.w * s);
    }

    const int wave  = tid >> 5;
    const int lane  = tid & 31;
    const int waveM = wave >> 2;    // 0..1  -> 64 rows each
    const int waveN = wave & 3;     // 0..3  -> 32 cols each
    const int lhi   = lane >> 4;    // K-half selector (ISA 16-bit A/B layout)
    const int llo   = lane & 15;    // row/col within fragment

    // Dynamic-LDS-only kernel: wave LDS offset of As region is the raw byte
    // offset (needed because async-to-LDS takes the LDS address in a VGPR).
    const unsigned AS_OFF = (unsigned)(NTILE * KPAD * 2); // 133120

    for (int mblk = 0; mblk < 4; ++mblk) {
        const int m0g = mblk * MTILE;
        __syncthreads();            // B ready / previous iter done with LDS

        // ---- stage A via CDNA5 async copy: bf16 ex tile -> LDS (16B chunks)
        // 128 rows * 1024B = 8192 chunks / 256 threads = 32 per thread.
        for (int c = tid; c < MTILE * 64; c += 256) {
            const int m  = c >> 6;          // row (64 x 16B chunks per row)
            const int kc = c & 63;          // 16B chunk within row
            const unsigned long long gaddr =
                (unsigned long long)(uintptr_t)(exb +
                    (((size_t)(m0g + m)) << 9) + ((size_t)kc << 3));
            const unsigned ldsa = AS_OFF + (unsigned)(m * (KPAD * 2) + kc * 16);
            asm volatile("global_load_async_to_lds_b128 %0, %1, off"
                         :: "v"(ldsa), "v"(gaddr) : "memory");
        }
        if (tid < MTILE) {
            sAt[tid] = a_t[m0g + tid];
            sLb[tid] = labels[m0g + tid];
        }
        asm volatile("s_wait_asynccnt 0x0" ::: "memory");
        __syncthreads();

        // ---- compute: 4(M) x 2(N) accumulator tiles per wave ----
        v8f acc[4][2];
#pragma unroll
        for (int tm = 0; tm < 4; ++tm)
#pragma unroll
            for (int tn = 0; tn < 2; ++tn)
#pragma unroll
                for (int e = 0; e < 8; ++e) acc[tm][tn][e] = 0.f;

        for (int k0 = 0; k0 < D_DIM; k0 += 32) {
            v16bf afr[4];
#pragma unroll
            for (int tm = 0; tm < 4; ++tm) {
                const bf16_t* base =
                    As + (waveM * 64 + tm * 16 + llo) * KPAD + k0 + lhi * 8;
                union { v16bf v; v8bf h[2]; } u;
                u.h[0] = *(const v8bf*)(base);       // K = k0 + lhi*8 .. +7
                u.h[1] = *(const v8bf*)(base + 16);  // K = k0+16+lhi*8 .. +7
                afr[tm] = u.v;
            }
            v16bf bfr[2];
#pragma unroll
            for (int tn = 0; tn < 2; ++tn) {
                // B^T columns are weight rows: symmetric fragment pattern
                const bf16_t* base =
                    Bs + (waveN * 32 + tn * 16 + llo) * KPAD + k0 + lhi * 8;
                union { v16bf v; v8bf h[2]; } u;
                u.h[0] = *(const v8bf*)(base);
                u.h[1] = *(const v8bf*)(base + 16);
                bfr[tn] = u.v;
            }
#pragma unroll
            for (int tm = 0; tm < 4; ++tm)
#pragma unroll
                for (int tn = 0; tn < 2; ++tn)
                    acc[tm][tn] = __builtin_amdgcn_wmma_f32_16x16x32_bf16(
                        false, afr[tm], false, bfr[tn],
                        (short)0, acc[tm][tn], false, false);
        }

        // ---- fused epilogue (branchless select; guard only on last tile) ----
        if (full_tile)
            arc_epilogue<false>(acc, sAt, sLb, n0, waveM, waveN, lhi, llo,
                                m0g, out);
        else
            arc_epilogue<true>(acc, sAt, sLb, n0, waveM, waveN, lhi, llo,
                               m0g, out);
    }
}

// ---------------------------------------------------------------------------
extern "C" void kernel_launch(void* const* d_in, const int* in_sizes, int n_in,
                              void* d_out, int out_size, void* d_ws, size_t ws_size,
                              hipStream_t stream) {
    (void)in_sizes; (void)n_in; (void)out_size; (void)ws_size;
    const float* feats  = (const float*)d_in[0];   // [512,512] f32
    const float* weight = (const float*)d_in[1];   // [100000,512] f32
    const int*   labels = (const int*)d_in[2];     // [512] int
    float*       out    = (float*)d_out;           // [512,100000] f32

    char* ws = (char*)d_ws;
    float*  invw = (float*)ws;                         // C floats   (400000 B)
    float*  at   = (float*)(ws + 400000);              // 512 floats (2048 B)
    bf16_t* exb  = (bf16_t*)(ws + 400000 + 2048);      // 512*512 bf16 (512 KB)

    wnorm_k<<<(C_DIM + 7) / 8, 256, 0, stream>>>(weight, invw);
    fprep_k<<<B_DIM, 256, 0, stream>>>(feats, weight, labels, invw, exb, at);

    const size_t lds = (size_t)(NTILE + MTILE) * KPAD * 2  // bf16 tiles
                     + (size_t)MTILE * 4                   // a_t
                     + (size_t)MTILE * 4;                  // labels
    arcneg_gemm_k<<<(C_DIM + NTILE - 1) / NTILE, 256, lds, stream>>>(
        weight, invw, exb, at, labels, out);
}
